// easymamba_model_26542897889398
// MI455X (gfx1250) — matrix operand order repaired and checked by reading, not compiled, not run on hardware
//
#include <hip/hip_runtime.h>
#include <math.h>

typedef __attribute__((ext_vector_type(16))) _Float16 v16h;
typedef __attribute__((ext_vector_type(8)))  _Float16 v8h;
typedef __attribute__((ext_vector_type(8)))  float    v8f;
typedef __attribute__((ext_vector_type(4)))  float    v4f;

#define NB   8        // batch
#define SEQ  256      // sequence length after conv stack
#define DM   512      // d_model
#define DI   1024     // d_inner
#define DS   16       // d_state
#define DTR  32       // dt_rank
#define NTOK (NB*SEQ) // 2048 tokens

static __device__ __forceinline__ float siluf(float x){ return x / (1.f + __expf(-x)); }
static __device__ __forceinline__ float softplusf(float x){ return x > 20.f ? x : log1pf(__expf(x)); }

// ---------------- front convs (conv0..conv3), fp32 direct ----------------
__global__ void k_conv_direct(const float* __restrict__ in, const float* __restrict__ w,
                              const float* __restrict__ bias, float* __restrict__ out,
                              int Cin, int Lin, int Cout, int Lout, int stride)
{
  int idx = blockIdx.x*blockDim.x + threadIdx.x;
  int total = NB*Cout*Lout;
  if (idx >= total) return;
  int t  = idx % Lout;
  int co = (idx / Lout) % Cout;
  int b  = idx / (Lout*Cout);
  float acc = bias[co];
  int p0 = t*stride - 7;
  const float* wr = w + (size_t)co*Cin*15;
  const float* ib = in + (size_t)b*Cin*Lin;
  for (int ci=0; ci<Cin; ++ci) {
    const float* ip = ib + (size_t)ci*Lin;
    const float* wk = wr + ci*15;
#pragma unroll
    for (int k=0;k<15;++k) {
      int p = p0 + k;
      float xv = (p>=0 && p<Lin) ? ip[p] : 0.f;
      acc += wk[k]*xv;
    }
  }
  out[idx] = acc;
}

// ---------------- fp32 -> f16 convert ----------------
__global__ void k_f32_to_f16(const float* __restrict__ s, _Float16* __restrict__ d, int n){
  int i = blockIdx.x*blockDim.x + threadIdx.x;
  if (i<n) d[i] = (_Float16)s[i];
}

// ---------------- im2col (f16 output, token-major rows of Cin*15) ----------------
__global__ void k_im2col_f16(const float* __restrict__ src, _Float16* __restrict__ dst,
                             int Cin, int Lin, int Lout, int stride, int token_major)
{
  int idx = blockIdx.x*blockDim.x + threadIdx.x;
  int total = NB*Lout*Cin*15;
  if (idx>=total) return;
  int k  = idx % 15;
  int ci = (idx/15) % Cin;
  int t  = (idx/(15*Cin)) % Lout;
  int b  = idx/(15*Cin*Lout);
  int p = t*stride - 7 + k;
  float v = 0.f;
  if (p>=0 && p<Lin)
    v = token_major ? src[((size_t)b*Lin + p)*Cin + ci]
                    : src[((size_t)b*Cin + ci)*Lin + p];
  dst[idx] = (_Float16)v;
}

// ---------------- register-blocked WMMA GEMM ----------------
// C[n][m] = bias[m] + sum_k A[m][k]*Bt[n][k]    (reference's  act @ W.T)
// A: MxK row-major f16 (weights).  Bt: NxK row-major f16 (token-major activations).
// C: NxM fp32 (ldc).  Requires: M,N multiples of 64; K multiple of 32; lda/ldb mult of 8.
// One wave owns a 64x64 tile: 4 A-frags x 4 B-frags -> 16 v_wmma per k-step
// (~1.0 global_load_b128 per WMMA). __launch_bounds__(256,1) lifts the 128-VGPR cap.
// Epilogue is compile-time specialized (ACT/HAS_BIAS/HAS_CH) and vector-wide
// (v4f/v8h stores on 32B-aligned rows) to keep accumulators in registers.
// ACT: 0=none, 1=softplus, 2=accumulate into C (fused residual add).
template<int ACT, bool HAS_BIAS, bool HAS_CH>
__global__ __launch_bounds__(256, 1)
void k_gemm_wmma(const _Float16* __restrict__ A, int lda,
                 const _Float16* __restrict__ Bt, int ldb,
                 const float* __restrict__ bias,
                 float* __restrict__ C, int ldc,
                 _Float16* __restrict__ Ch,
                 int M, int N, int K)
{
  int wave = blockIdx.x*(blockDim.x>>5) + (threadIdx.x>>5);
  int tN = N>>6, tM = M>>6;
  if (wave >= tM*tN) return;                 // wave-uniform: EXEC all-1s for WMMA
  int m0 = (wave / tN) << 6;
  int n0 = (wave % tN) << 6;
  int lane = threadIdx.x & 31;
  int l15 = lane & 15, hi = lane>>4;
  int ab = hi<<3;   // A lane K-base: 0 / 8
  int bb = hi<<4;   // 0 / 16, unused (B loads use ab)
  const _Float16* arow[4]; const _Float16* brow[4];
#pragma unroll
  for (int i=0;i<4;++i){
    arow[i] = A  + (size_t)(m0 + i*16 + l15)*lda;
    brow[i] = Bt + (size_t)(n0 + i*16 + l15)*ldb;
  }
  v8f acc[4][4];
#pragma unroll
  for (int i=0;i<4;++i)
#pragma unroll
    for (int j=0;j<4;++j) acc[i][j] = (v8f){0.f,0.f,0.f,0.f,0.f,0.f,0.f,0.f};

  for (int k0=0;k0<K;k0+=32){
    v16h av[4], bv[4];
#pragma unroll
    for (int i=0;i<4;++i){
      union { v16h v; v8h h[2]; } u;
      // A 16x32 layout: e<8 -> K=ab+e ; e>=8 -> K=16+ab+(e-8)
      u.h[0] = *(const v8h*)(arow[i] + k0 + ab);
      u.h[1] = *(const v8h*)(arow[i] + k0 + 16 + ab);
      av[i] = u.v;
    }
#pragma unroll
    for (int j=0;j<4;++j){
      union { v16h v; v8h h[2]; } u;
      // B 32x16 layout: e<8 -> K=ab+e ; e>=8 -> K=16+ab+(e-8)
      u.h[0] = *(const v8h*)(brow[j] + k0 + ab);
      u.h[1] = *(const v8h*)(brow[j] + k0 + 16 + ab);
      bv[j] = u.v;
    }
#pragma unroll
    for (int i=0;i<4;++i)
#pragma unroll
      for (int j=0;j<4;++j)
        acc[i][j] = __builtin_amdgcn_wmma_f32_16x16x32_f16(false, av[i], false, bv[j],
                                                           (short)0, acc[i][j], false, false);
  }

#pragma unroll
  for (int j=0;j<4;++j){
    int n = n0 + j*16 + l15;
    float*    cbase  = C + (size_t)n*ldc;
#pragma unroll
    for (int i=0;i<4;++i){
      int mb = m0 + i*16 + (hi<<3);          // C/D: element r -> M = mb + r; mb mult of 8
      v8f v = acc[i][j];
      if (HAS_BIAS){
        v4f blo = *(const v4f*)(bias + mb);
        v4f bhi = *(const v4f*)(bias + mb + 4);
#pragma unroll
        for (int r=0;r<4;++r){ v[r] += blo[r]; v[r+4] += bhi[r]; }
      }
      if (ACT==1){
#pragma unroll
        for (int r=0;r<8;++r) v[r] = softplusf(v[r]);
      }
      float* crow = cbase + mb;
      if (ACT==2){
        v4f clo = *(const v4f*)crow;
        v4f chi = *(const v4f*)(crow + 4);
#pragma unroll
        for (int r=0;r<4;++r){ v[r] += clo[r]; v[r+4] += chi[r]; }
      }
      *(v4f*)(crow)     = (v4f){v[0],v[1],v[2],v[3]};
      *(v4f*)(crow + 4) = (v4f){v[4],v[5],v[6],v[7]};
      if (HAS_CH){
        v8h hv;
#pragma unroll
        for (int r=0;r<8;++r) hv[r] = (_Float16)v[r];
        *(v8h*)(Ch + (size_t)n*ldc + mb) = hv;   // 16B aligned (ldc mult 8, mb mult 8)
      }
    }
  }
}

// ---------------- RMSNorm (per token) + f16 emit ----------------
__global__ void k_rmsnorm_f16(const float* __restrict__ h, const float* __restrict__ w,
                              _Float16* __restrict__ out)
{
  __shared__ float red[256];
  int tok = blockIdx.x;
  const float* x = h + (size_t)tok*DM;
  float s = 0.f;
  for (int d=threadIdx.x; d<DM; d+=blockDim.x) { float v = x[d]; s += v*v; }
  red[threadIdx.x] = s; __syncthreads();
  for (int off=128; off>0; off>>=1){
    if ((int)threadIdx.x < off) red[threadIdx.x] += red[threadIdx.x+off];
    __syncthreads();
  }
  float rs = rsqrtf(red[0]*(1.f/DM) + 1e-5f);
  for (int d=threadIdx.x; d<DM; d+=blockDim.x)
    out[(size_t)tok*DM + d] = (_Float16)(x[d]*rs*w[d]);
}

// ---------------- depthwise causal conv1d (k=3) + bias + SiLU ----------------
__global__ void k_dwconv_silu(const float* __restrict__ xz, const float* __restrict__ w,
                              const float* __restrict__ bias, float* __restrict__ xbc,
                              _Float16* __restrict__ xbc_h)
{
  int idx = blockIdx.x*blockDim.x + threadIdx.x;
  if (idx >= NTOK*DI) return;
  int d = idx % DI;
  int t = (idx / DI) % SEQ;
  int b = idx / (DI*SEQ);
  float acc = bias[d];
#pragma unroll
  for (int j=0;j<3;++j){
    int ts = t-2+j;
    if (ts>=0) acc += w[d*3+j]*xz[(size_t)(b*SEQ+ts)*(2*DI) + d];
  }
  acc = siluf(acc);
  xbc[idx]   = acc;
  xbc_h[idx] = (_Float16)acc;
}

// ---------------- fused selective scan: one lane per (b,d,state) ----------------
// 131072 threads; y = sum_n h[n]*C[n] via 16-lane xor-shuffle tree (wave32, masks 1/2/4/8
// stay inside each 16-lane half). Fuses +D*u and SiLU(z) gating; emits f16 for out_proj.
__global__ void k_scan(const float* __restrict__ delta, const float* __restrict__ dbl,
                       const float* __restrict__ xbc, const float* __restrict__ xz,
                       const float* __restrict__ A_log, const float* __restrict__ Dp,
                       _Float16* __restrict__ y_h)
{
  int idx = blockIdx.x*blockDim.x + threadIdx.x;
  if (idx >= NB*DI*DS) return;
  int n = idx & (DS-1);
  int g = idx >> 4;          // (b,d) group
  int d = g % DI;
  int b = g / DI;
  float a  = -__expf(A_log[d*DS + n]);
  float hs = 0.f;
  float Dd = Dp[d];
  for (int s=0;s<SEQ;++s){
    int tok = b*SEQ + s;
    float dt = delta[(size_t)tok*DI + d];
    float u  = xbc[(size_t)tok*DI + d];
    const float* bc = dbl + (size_t)tok*64;     // [0:32)=dt_in, [32:48)=B, [48:64)=C
    hs = __expf(dt*a)*hs + dt*u*bc[DTR + n];
    float y = hs * bc[DTR + DS + n];
    y += __shfl_xor(y, 1, 32);
    y += __shfl_xor(y, 2, 32);
    y += __shfl_xor(y, 4, 32);
    y += __shfl_xor(y, 8, 32);
    if (n == 0){
      y += Dd*u;
      float z = xz[(size_t)tok*(2*DI) + DI + d];
      y *= siluf(z);
      y_h[(size_t)tok*DI + d] = (_Float16)y;
    }
  }
}

// ---------------- mean pool over seq + FC (10 classes) ----------------
__global__ void k_pool_fc(const float* __restrict__ h, const float* __restrict__ fw,
                          const float* __restrict__ fb, float* __restrict__ out)
{
  __shared__ float pooled[DM];
  int b = blockIdx.x;
  for (int d=threadIdx.x; d<DM; d+=blockDim.x){
    float s=0.f;
    for (int t=0;t<SEQ;++t) s += h[(size_t)(b*SEQ+t)*DM + d];
    pooled[d] = s * (1.f/SEQ);
  }
  __syncthreads();
  if (threadIdx.x < 10){
    float s = fb[threadIdx.x];
    const float* wr = fw + threadIdx.x*DM;
    for (int d=0; d<DM; ++d) s += pooled[d]*wr[d];
    out[b*10 + threadIdx.x] = s;
  }
}

extern "C" void kernel_launch(void* const* d_in, const int* in_sizes, int n_in,
                              void* d_out, int out_size, void* d_ws, size_t ws_size,
                              hipStream_t stream)
{
  const float* x = (const float*)d_in[0];
  const float* cw[6]; const float* cb[6];
  for (int i=0;i<6;++i){ cw[i]=(const float*)d_in[1+2*i]; cb[i]=(const float*)d_in[2+2*i]; }
  const float* norm_w  = (const float*)d_in[13];
  const float* in_w    = (const float*)d_in[14];
  const float* c1d_w   = (const float*)d_in[15];
  const float* c1d_b   = (const float*)d_in[16];
  const float* xp_w    = (const float*)d_in[17];
  const float* dt_w    = (const float*)d_in[18];
  const float* dt_b    = (const float*)d_in[19];
  const float* A_log   = (const float*)d_in[20];
  const float* D_param = (const float*)d_in[21];
  const float* out_w   = (const float*)d_in[22];
  const float* fc_w    = (const float*)d_in[23];
  const float* fc_b    = (const float*)d_in[24];

  char* p = (char*)d_ws;
  auto alloc = [&](size_t bytes)->void* { void* r = (void*)p; p += (bytes + 255) & ~(size_t)255; return r; };

  float*    out0  = (float*)   alloc((size_t)8*16*8192*4);
  float*    out1  = (float*)   alloc((size_t)8*32*4096*4);
  float*    out2  = (float*)   alloc((size_t)8*64*2048*4);
  float*    out3  = (float*)   alloc((size_t)8*128*1024*4);
  _Float16* col   = (_Float16*)alloc((size_t)7864320*2);   // shared im2col (conv4 4096x1920 == conv5 2048x3840)
  float*    out4  = (float*)   alloc((size_t)4096*256*4);  // conv4 out, token-major
  float*    hbuf  = (float*)   alloc((size_t)NTOK*DM*4);   // residual stream (B,S,D)
  _Float16* w4h   = (_Float16*)alloc((size_t)491520*2);
  _Float16* w5h   = (_Float16*)alloc((size_t)1966080*2);
  _Float16* inwh  = (_Float16*)alloc((size_t)4194304*2);
  _Float16* xpwh  = (_Float16*)alloc((size_t)262144*2);
  _Float16* dtwh  = (_Float16*)alloc((size_t)131072*2);
  _Float16* outwh = (_Float16*)alloc((size_t)2097152*2);
  _Float16* hn_h  = (_Float16*)alloc((size_t)NTOK*DM*2);
  float*    xz    = (float*)   alloc((size_t)NTOK*2*DI*4);
  float*    xbc   = (float*)   alloc((size_t)NTOK*DI*4);
  _Float16* xbc_h = (_Float16*)alloc((size_t)NTOK*DI*2);
  float*    dbl   = (float*)   alloc((size_t)NTOK*64*4);
  _Float16* dbl_h = (_Float16*)alloc((size_t)NTOK*64*2);
  float*    delta = (float*)   alloc((size_t)NTOK*DI*4);
  _Float16* y_h   = (_Float16*)alloc((size_t)NTOK*DI*2);

  dim3 blk(256);
  auto grid1 = [](int n){ return dim3((unsigned)((n+255)/256)); };
  auto gemmGrid = [](int M,int N){ int waves=(M>>6)*(N>>6); return dim3((unsigned)((waves+7)/8)); };

  // weight conversions to f16
  k_f32_to_f16<<<grid1(491520), blk,0,stream>>>(cw[4], w4h, 491520);
  k_f32_to_f16<<<grid1(1966080),blk,0,stream>>>(cw[5], w5h, 1966080);
  k_f32_to_f16<<<grid1(4194304),blk,0,stream>>>(in_w,  inwh, 4194304);
  k_f32_to_f16<<<grid1(262144), blk,0,stream>>>(xp_w,  xpwh, 262144);
  k_f32_to_f16<<<grid1(131072), blk,0,stream>>>(dt_w,  dtwh, 131072);
  k_f32_to_f16<<<grid1(2097152),blk,0,stream>>>(out_w, outwh, 2097152);

  // conv0..conv3: direct fp32
  k_conv_direct<<<grid1(8*16*8192), blk,0,stream>>>(x,    cw[0], cb[0], out0, 16, 32768, 16, 8192, 4);
  k_conv_direct<<<grid1(8*32*4096), blk,0,stream>>>(out0, cw[1], cb[1], out1, 16, 8192,  32, 4096, 2);
  k_conv_direct<<<grid1(8*64*2048), blk,0,stream>>>(out1, cw[2], cb[2], out2, 32, 4096,  64, 2048, 2);
  k_conv_direct<<<grid1(8*128*1024),blk,0,stream>>>(out2, cw[3], cb[3], out3, 64, 2048, 128, 1024, 2);

  // conv4: im2col + WMMA GEMM  (M=256, N=4096 tokens, K=1920)
  k_im2col_f16<<<grid1(8*512*128*15),blk,0,stream>>>(out3, col, 128, 1024, 512, 2, 0);
  k_gemm_wmma<0,true,false><<<gemmGrid(256,4096),blk,0,stream>>>(w4h,1920, col,1920, cb[4], out4,256, nullptr, 256,4096,1920);

  // conv5: im2col + WMMA GEMM  (M=512, N=2048, K=3840) -> lands directly as h[b][t][d]
  k_im2col_f16<<<grid1(8*256*256*15),blk,0,stream>>>(out4, col, 256, 512, 256, 2, 1);
  k_gemm_wmma<0,true,false><<<gemmGrid(512,2048),blk,0,stream>>>(w5h,3840, col,3840, cb[5], hbuf,512, nullptr, 512,2048,3840);

  // 4 Mamba layers
  for (int l=0;l<4;++l){
    k_rmsnorm_f16<<<dim3(NTOK),blk,0,stream>>>(hbuf, norm_w + (size_t)l*DM, hn_h);
    // in_proj: M=2048, K=512
    k_gemm_wmma<0,false,false><<<gemmGrid(2048,NTOK),blk,0,stream>>>(inwh + (size_t)l*2048*512, 512,
                                                                     hn_h, 512, nullptr, xz, 2048, nullptr,
                                                                     2048, NTOK, 512);
    k_dwconv_silu<<<grid1(NTOK*DI),blk,0,stream>>>(xz, c1d_w + (size_t)l*DI*3, c1d_b + (size_t)l*DI, xbc, xbc_h);
    // x_proj: M=64, K=1024; epilogue also emits f16 mirror for dt_proj input
    k_gemm_wmma<0,false,true><<<gemmGrid(64,NTOK),blk,0,stream>>>(xpwh + (size_t)l*64*1024, 1024,
                                                                  xbc_h, 1024, nullptr, dbl, 64, dbl_h,
                                                                  64, NTOK, 1024);
    // dt_proj: M=1024, K=32 (first 32 cols of dbl rows), fused bias+softplus
    k_gemm_wmma<1,true,false><<<gemmGrid(1024,NTOK),blk,0,stream>>>(dtwh + (size_t)l*1024*32, 32,
                                                                    dbl_h, 64, dt_b + (size_t)l*DI, delta, 1024, nullptr,
                                                                    1024, NTOK, 32);
    k_scan<<<grid1(NB*DI*DS),blk,0,stream>>>(delta, dbl, xbc, xz,
                                             A_log + (size_t)l*DI*DS, D_param + (size_t)l*DI, y_h);
    // out_proj: M=512, K=1024; ACT=2 -> fused residual accumulate into hbuf
    k_gemm_wmma<2,false,false><<<gemmGrid(512,NTOK),blk,0,stream>>>(outwh + (size_t)l*512*1024, 1024,
                                                                    y_h, 1024, nullptr, hbuf, 512, nullptr,
                                                                    512, NTOK, 1024);
  }

  k_pool_fc<<<dim3(NB),blk,0,stream>>>(hbuf, fc_w, fc_b, (float*)d_out);

  (void)in_sizes; (void)n_in; (void)out_size; (void)ws_size;
}
